// ColBERT_60936995995765
// MI455X (gfx1250) — compile-verified, matrix operand order checked by reading, not run on hardware
//
#include <hip/hip_runtime.h>
#include <hip/hip_bf16.h>

// ---------------- WMMA types (CDNA5 wave32) ----------------
typedef __attribute__((ext_vector_type(16))) __bf16 v16bf;
typedef __attribute__((ext_vector_type(8)))  float  v8f;
typedef __attribute__((ext_vector_type(4)))  float  v4f;   // clang ext vector: OK for NT builtins

union BF16x16 { v16bf v; unsigned u[8]; };

#define A_STRIDE  772   // fp32 staging row stride (768 + 4 pad)
#define AB_STRIDE 776   // bf16 tile row stride in elements (768 + 8 pad; 16B-aligned rows)
#define AB_U      (AB_STRIDE / 2)
#define T_STRIDE  136   // doc-tile bf16 row stride (128 + 8 pad)

__device__ __forceinline__ unsigned short f2bf(float a) {          // RNE, cold paths only
    unsigned u = __float_as_uint(a);
    return (unsigned short)((u + 0x7FFFu + ((u >> 16) & 1u)) >> 16);
}
// pack two fp32 -> two bf16 (truncate) in ONE v_perm_b32
__device__ __forceinline__ unsigned pk2bf_trunc(float x, float y) {
    return __builtin_amdgcn_perm(__float_as_uint(y), __float_as_uint(x), 0x07060302u);
}
// monotone order-preserving float<->uint for atomicMax
__device__ __forceinline__ unsigned fenc(float f) {
    unsigned u = __float_as_uint(f);
    return (u >> 31) ? ~u : (u | 0x80000000u);
}
__device__ __forceinline__ float fdec(unsigned e) {
    unsigned u = (e >> 31) ? (e ^ 0x80000000u) : ~e;
    return __uint_as_float(u);
}

// 16-lane XOR butterfly reductions via ds_swizzle (1 instruction per step, no bounds checks).
// pattern = {0, xor[14:10], or[9:5]=0, and[4:0]=0x1F}: src_lane = lane ^ m (stays in 16-lane half)
__device__ __forceinline__ float redsum16(float s) {
    s += __int_as_float(__builtin_amdgcn_ds_swizzle(__float_as_int(s), 0x041F));
    s += __int_as_float(__builtin_amdgcn_ds_swizzle(__float_as_int(s), 0x081F));
    s += __int_as_float(__builtin_amdgcn_ds_swizzle(__float_as_int(s), 0x101F));
    s += __int_as_float(__builtin_amdgcn_ds_swizzle(__float_as_int(s), 0x201F));
    return s;
}
__device__ __forceinline__ float redmax16(float s) {
    s = fmaxf(s, __int_as_float(__builtin_amdgcn_ds_swizzle(__float_as_int(s), 0x041F)));
    s = fmaxf(s, __int_as_float(__builtin_amdgcn_ds_swizzle(__float_as_int(s), 0x081F)));
    s = fmaxf(s, __int_as_float(__builtin_amdgcn_ds_swizzle(__float_as_int(s), 0x101F)));
    s = fmaxf(s, __int_as_float(__builtin_amdgcn_ds_swizzle(__float_as_int(s), 0x201F)));
    return s;
}

// 16-bit A/B fragment from contiguous bf16: p -> element [row_or_col][khalf*8] of the k-chunk.
// v0..3 hold k = khalf*8 + 0..7 ; v4..7 hold k = 16 + khalf*8 + 0..7 (ISA 7.12.2).
__device__ __forceinline__ v16bf frag_from_bf16(const unsigned short* p) {
    uint4 a = *(const uint4*)(p);       // k + 0..7
    uint4 b = *(const uint4*)(p + 16);  // k + 16..23
    BF16x16 r;
    r.u[0] = a.x; r.u[1] = a.y; r.u[2] = a.z; r.u[3] = a.w;
    r.u[4] = b.x; r.u[5] = b.y; r.u[6] = b.z; r.u[7] = b.w;
    return r.v;
}

__device__ __forceinline__ v8f wmma_bf16(v16bf a, v16bf b, v8f c) {
    return __builtin_amdgcn_wmma_f32_16x16x32_bf16(false, a, false, b, (short)0, c, false, false);
}

// ---- async staging (kept on the q path): CDNA5 global->LDS DMA, non-temporal ----
__device__ __forceinline__ void stage_tile_async(const float* __restrict__ src, float* A_lds, int tid) {
#pragma unroll
    for (int i = 0; i < 12; ++i) {
        int idx = i * 256 + tid;              // 3072 16-byte chunks
        int r = idx / 192, ch = idx % 192;
        unsigned lds_off = (unsigned)(uintptr_t)&A_lds[r * A_STRIDE + ch * 4];
        unsigned long long g = (unsigned long long)(uintptr_t)(src + r * 768 + ch * 4);
        asm volatile("global_load_async_to_lds_b128 %0, %1, off th:TH_LOAD_NT"
                     :: "v"(lds_off), "v"(g) : "memory");
    }
    asm volatile("s_wait_asynccnt 0x0" ::: "memory");
}
__device__ __forceinline__ void cvt_tile_bf16(const float* A_lds, unsigned* A16u, int tid) {
#pragma unroll
    for (int i = 0; i < 24; ++i) {
        int idx = i * 256 + tid;              // 6144 packed uints
        int r = idx / 384, kk = idx % 384;
        float2 f = *(const float2*)&A_lds[r * A_STRIDE + kk * 2];
        A16u[r * AB_U + kk] = pk2bf_trunc(f.x, f.y);
    }
}

// ---- direct staging (doc path): NT global b128 -> 2x v_perm -> ds_store_b64 ----
__device__ __forceinline__ void stage_tile_direct(const float* __restrict__ src, unsigned* dst, int tid) {
#pragma unroll
    for (int i = 0; i < 12; ++i) {
        int idx = i * 256 + tid;              // 3072 16-byte chunks
        int r = idx / 192, ch = idx % 192;
        v4f f = __builtin_nontemporal_load((const v4f*)(src + r * 768 + ch * 4));
        uint2 p;
        p.x = pk2bf_trunc(f[0], f[1]);
        p.y = pk2bf_trunc(f[2], f[3]);
        *(uint2*)&dst[r * AB_U + ch * 2] = p;
    }
}

// ---------------- kernel 1: Wt bf16 transpose + score-buffer init ----------------
__global__ __launch_bounds__(256) void prep_kernel(const float* __restrict__ W,
                                                   unsigned short* __restrict__ Wt,
                                                   unsigned* __restrict__ posE,
                                                   unsigned* __restrict__ negE) {
    int i = blockIdx.x * 256 + threadIdx.x;          // 98304 elements of Wt[d][k]
    if (i < 768 * 128) {
        int d = i / 768, k = i % 768;
        Wt[i] = f2bf(W[k * 128 + d]);
    }
    if (i < 4096) { posE[i] = 0u; negE[i] = 0u; }
}

// ---------------- kernel 2: Q = l2norm(q_hidden @ W + b), store bf16 ----------------
__global__ __launch_bounds__(256) void qproj_kernel(const float* __restrict__ qh,
                                                    const unsigned short* __restrict__ Wt,
                                                    const float* __restrict__ bias,
                                                    unsigned short* __restrict__ Qb) {
    __shared__ float    A_lds[16 * A_STRIDE];
    __shared__ unsigned A16u[16 * AB_U];
    __shared__ float    rowsq[16];
    int tid = threadIdx.x, tile = blockIdx.x;        // 256 tiles of 16 q-rows
    if (tid < 16) rowsq[tid] = 0.f;
    stage_tile_async(qh + (size_t)tile * 16 * 768, A_lds, tid);
    __syncthreads();
    cvt_tile_bf16(A_lds, A16u, tid);
    __syncthreads();

    int wave = tid >> 5, lane = tid & 31, m = lane & 15, kh = lane >> 4;
    const unsigned short* ab = (const unsigned short*)A16u + m * AB_STRIDE + kh * 8;
    const unsigned short* bb = Wt + (size_t)(wave * 16 + m) * 768 + kh * 8;
    v8f c = {};
#pragma unroll 8
    for (int kc = 0; kc < 24; ++kc)
        c = wmma_bf16(frag_from_bf16(ab + kc * 32), frag_from_bf16(bb + kc * 32), c);

    int n = lane & 15, grp = lane >> 4;
    float bv = bias[wave * 16 + n];
#pragma unroll
    for (int v = 0; v < 8; ++v) {
        float x = c[v] + bv;
        c[v] = x;
        float s = redsum16(x * x);                    // row m = grp*8+v over this wave's 16 cols
        if (n == 0) atomicAdd(&rowsq[grp * 8 + v], s);
    }
    __syncthreads();
#pragma unroll
    for (int v = 0; v < 8; ++v) {
        int mm = grp * 8 + v;
        float inv = 1.f / fmaxf(sqrtf(rowsq[mm]), 1e-12f);
        Qb[(size_t)(tile * 16 + mm) * 128 + wave * 16 + n] = f2bf(c[v] * inv);
    }
}

// ---- kernel 3/4: 4-tile fused doc projection + mask + l2norm + MaxSim ----
// One workgroup = 64 doc tokens of one batch. B-fragments register-reused across the
// 4 tiles (4x less Wt/L2 traffic); MaxSim uses all 8 waves (2 q-halves x 4 token-tiles).
__global__ __launch_bounds__(256) void docscore_kernel(const float* __restrict__ dh,
                                                       const int* __restrict__ mask,
                                                       const unsigned short* __restrict__ Wt,
                                                       const float* __restrict__ bias,
                                                       const unsigned short* __restrict__ Qb,
                                                       unsigned* __restrict__ scoreE) {
    __shared__ unsigned       A16u[4][16 * AB_U];       // 4 bf16 tiles, 99 KB
    __shared__ float          rowsq[64];
    __shared__ float          rowmask[64];
    __shared__ unsigned short T_lds[4][16 * T_STRIDE];  // normalized doc tiles, 17 KB
    int tid = threadIdx.x, g = blockIdx.x;              // 1024 groups: b = g>>3, 64 tokens each
    int b = g >> 3, tok0 = (g & 7) * 64;
    const float* src = dh + ((size_t)b * 512 + tok0) * 768;
    if (tid < 64) { rowsq[tid] = 0.f; rowmask[tid] = (float)mask[b * 512 + tok0 + tid]; }
#pragma unroll
    for (int t = 0; t < 4; ++t)
        stage_tile_direct(src + t * 16 * 768, A16u[t], tid);
    __syncthreads();

    int wave = tid >> 5, lane = tid & 31, m = lane & 15, kh = lane >> 4;
    int n = lane & 15, grp = lane >> 4;
    const unsigned short* ab = (const unsigned short*)A16u[0] + m * AB_STRIDE + kh * 8;
    const unsigned short* bb = Wt + (size_t)(wave * 16 + m) * 768 + kh * 8;
    const int AT = 16 * AB_STRIDE;                      // bf16 elements per tile buffer
    v8f acc[4] = {{}, {}, {}, {}};
#pragma unroll 4
    for (int kc = 0; kc < 24; ++kc) {
        v16bf bfr = frag_from_bf16(bb + kc * 32);       // loaded once, used 4x
        acc[0] = wmma_bf16(frag_from_bf16(ab + 0 * AT + kc * 32), bfr, acc[0]);
        acc[1] = wmma_bf16(frag_from_bf16(ab + 1 * AT + kc * 32), bfr, acc[1]);
        acc[2] = wmma_bf16(frag_from_bf16(ab + 2 * AT + kc * 32), bfr, acc[2]);
        acc[3] = wmma_bf16(frag_from_bf16(ab + 3 * AT + kc * 32), bfr, acc[3]);
    }

    float bv = bias[wave * 16 + n];
#pragma unroll
    for (int t = 0; t < 4; ++t) {
#pragma unroll
        for (int v = 0; v < 8; ++v) {
            int row = t * 16 + grp * 8 + v;
            float x = (acc[t][v] + bv) * rowmask[row];  // mask BEFORE normalize (ref semantics)
            acc[t][v] = x;
            float s = redsum16(x * x);
            if (n == 0) atomicAdd(&rowsq[row], s);
        }
    }
    __syncthreads();
#pragma unroll
    for (int t = 0; t < 4; ++t) {
#pragma unroll
        for (int v = 0; v < 8; ++v) {
            int row = t * 16 + grp * 8 + v;
            float inv = 1.f / fmaxf(sqrtf(rowsq[row]), 1e-12f);
            T_lds[t][(grp * 8 + v) * T_STRIDE + wave * 16 + n] = f2bf(acc[t][v] * inv);
        }
    }
    __syncthreads();

    // MaxSim: S(32q x 64tok); wave w -> token-tile w>>1, q-half w&1. All 8 waves WMMA.
    {
        int tt = wave >> 1, qhalf = wave & 1;
        const unsigned short* qbase = Qb + ((size_t)(b * 32 + qhalf * 16 + m)) * 128 + kh * 8;
        const unsigned short* tbase = T_lds[tt] + m * T_STRIDE + kh * 8;
        v8f s = {};
#pragma unroll
        for (int kc = 0; kc < 4; ++kc)                  // K = D = 128 in 4 chunks of 32
            s = wmma_bf16(frag_from_bf16(qbase + kc * 32), frag_from_bf16(tbase + kc * 32), s);
#pragma unroll
        for (int v = 0; v < 8; ++v) {
            float mx = redmax16(s[v]);                  // max over 16 token cols
            if (n == 0) atomicMax(&scoreE[b * 32 + qhalf * 16 + grp * 8 + v], fenc(mx));
        }
    }
}

// ---------------- kernel 5: decode + sum over q -> [B, 2] ----------------
__global__ __launch_bounds__(256) void final_kernel(const unsigned* __restrict__ posE,
                                                    const unsigned* __restrict__ negE,
                                                    float* __restrict__ out) {
    int t = threadIdx.x;                             // 256 = 128 batches x {pos,neg}
    int b = t >> 1;
    const unsigned* src = (t & 1) ? negE : posE;
    float s = 0.f;
#pragma unroll 8
    for (int q = 0; q < 32; ++q) s += fdec(src[b * 32 + q]);
    out[t] = s;                                      // out[b*2 + sel]
}

extern "C" void kernel_launch(void* const* d_in, const int* in_sizes, int n_in,
                              void* d_out, int out_size, void* d_ws, size_t ws_size,
                              hipStream_t stream) {
    const float* qh   = (const float*)d_in[0];
    const float* pdh  = (const float*)d_in[1];
    const float* ndh  = (const float*)d_in[2];
    const float* W    = (const float*)d_in[3];
    const float* bias = (const float*)d_in[4];
    const int*   pdm  = (const int*)d_in[5];
    const int*   ndm  = (const int*)d_in[6];

    char* ws = (char*)d_ws;
    unsigned short* Wt  = (unsigned short*)ws;                       // 196608 B  (Wt[d][k] bf16)
    unsigned short* Qb  = (unsigned short*)(ws + 196608);            // 1048576 B (Q bf16, normalized)
    unsigned*       posE = (unsigned*)(ws + 196608 + 1048576);       // 16384 B
    unsigned*       negE = posE + 4096;                              // 16384 B

    prep_kernel<<<384, 256, 0, stream>>>(W, Wt, posE, negE);
    qproj_kernel<<<256, 256, 0, stream>>>(qh, Wt, bias, Qb);
    docscore_kernel<<<1024, 256, 0, stream>>>(pdh, pdm, Wt, bias, Qb, posE);
    docscore_kernel<<<1024, 256, 0, stream>>>(ndh, ndm, Wt, bias, Qb, negE);
    final_kernel<<<1, 256, 0, stream>>>(posE, negE, (float*)d_out);
}
